// Model_5325759447710
// MI455X (gfx1250) — matrix operand order repaired and checked by reading, not compiled, not run on hardware
//
#include <hip/hip_runtime.h>
#include <hip/hip_bf16.h>

typedef __bf16 bf16;
typedef __attribute__((ext_vector_type(16))) __bf16 v16bf;
typedef __attribute__((ext_vector_type(8)))  float  v8f;

typedef int vint4 __attribute__((ext_vector_type(4)));
typedef __attribute__((address_space(1))) vint4 as1_vint4;   // global int4
typedef __attribute__((address_space(3))) vint4 as3_vint4;   // LDS int4

#define BB 4
#define NN 8192
#define LL 8
#define CC 256
#define HH 8
#define DD 32
#define GG 64
#define FF 1024
#define MT (BB*NN)   // 32768 rows

#if __has_builtin(__builtin_amdgcn_global_load_async_to_lds_b128)
#define HAVE_ASYNC 1
#else
#define HAVE_ASYNC 0
#endif

#if HAVE_ASYNC
#if __has_builtin(__builtin_amdgcn_s_wait_asynccnt)
#define WAIT_ASYNC(n) __builtin_amdgcn_s_wait_asynccnt(n)
#else
#define WAIT_ASYNC(n) asm volatile("s_wait_asynccnt %0" :: "i"(n))
#endif
#else
#define WAIT_ASYNC(n) ((void)0)
#endif

__device__ __forceinline__ float gelu_f(float x) {
    return 0.5f * x * (1.0f + erff(x * 0.70710678118654752f));
}

union Frag { v16bf v; uint4 q[2]; };

// ---------------------------------------------------------------------------
// Generic bf16 WMMA GEMM:  out = act( A[M,K] @ W[K,N] + bias ) (+ residual)
// W is pre-transposed: Wt[N,K].  Both LDS tiles are straight row-major copies
// of 64B rows -> staged with GLOBAL_LOAD_ASYNC_TO_LDS_B128, double buffered.
// Block tile 128x128, 8 waves, wave tile 32x64 (2x4 WMMA 16x16 tiles).
// ---------------------------------------------------------------------------
template<bool GELU, bool RES, bool WF, bool WB>
__global__ __launch_bounds__(256)
void gemm_bf16_kernel(const bf16* __restrict__ A, const bf16* __restrict__ Wt,
                      const float* __restrict__ bias,
                      const float* residual,
                      float* outf, bf16* outb,
                      int N, int K)
{
    __shared__ bf16 As[2][128 * 32];       // [buf][row][k]
    __shared__ bf16 Bs[2][128 * 32];       // [buf][col][k]

    const int tid  = threadIdx.x;
    const int lane = tid & 31;
    const int wave = tid >> 5;
    const int wm   = wave >> 1;            // 0..3
    const int wn   = wave & 1;             // 0..1
    const int row0 = blockIdx.y * 128;
    const int col0 = blockIdx.x * 128;

    v8f acc[2][4];
    #pragma unroll
    for (int mt = 0; mt < 2; ++mt)
        #pragma unroll
        for (int nt = 0; nt < 4; ++nt)
            acc[mt][nt] = (v8f){0.f,0.f,0.f,0.f,0.f,0.f,0.f,0.f};

    const int srow = tid >> 1;             // staging: row/col 0..127
    const int sseg = (tid & 1) * 16;       // staging: k-offset 0/16

    auto stage = [&](int k0, int buf) {
        const bf16* ga = A  + (size_t)(row0 + srow) * K + k0 + sseg;
        const bf16* gb = Wt + (size_t)(col0 + srow) * K + k0 + sseg;
        bf16* sa = &As[buf][srow * 32 + sseg];
        bf16* sb = &Bs[buf][srow * 32 + sseg];
#if HAVE_ASYNC
        __builtin_amdgcn_global_load_async_to_lds_b128(
            (as1_vint4*)(ga),     (as3_vint4*)(sa),     0, 0);
        __builtin_amdgcn_global_load_async_to_lds_b128(
            (as1_vint4*)(ga + 8), (as3_vint4*)(sa + 8), 0, 0);
        __builtin_amdgcn_global_load_async_to_lds_b128(
            (as1_vint4*)(gb),     (as3_vint4*)(sb),     0, 0);
        __builtin_amdgcn_global_load_async_to_lds_b128(
            (as1_vint4*)(gb + 8), (as3_vint4*)(sb + 8), 0, 0);
#else
        const uint4* ga4 = reinterpret_cast<const uint4*>(ga);
        const uint4* gb4 = reinterpret_cast<const uint4*>(gb);
        uint4* sa4 = reinterpret_cast<uint4*>(sa);
        uint4* sb4 = reinterpret_cast<uint4*>(sb);
        sa4[0] = ga4[0]; sa4[1] = ga4[1];
        sb4[0] = gb4[0]; sb4[1] = gb4[1];
#endif
    };

    const int r  = lane & 15;
    const int kh = (lane >> 4) * 8;        // A frag: K offset 0 or 8
    const int kb = (lane >> 4) * 16;       // unused; B frag uses kh (0 or 8) like A
    const int KS = K >> 5;

    stage(0, 0);
    for (int ks = 0; ks < KS; ++ks) {
        const int buf = ks & 1;
        if (ks + 1 < KS) {
            stage((ks + 1) << 5, buf ^ 1);
            WAIT_ASYNC(4);                 // current tile landed, next in flight
        } else {
            WAIT_ASYNC(0);
        }
        __syncthreads();

        Frag a[2];
        #pragma unroll
        for (int mt = 0; mt < 2; ++mt) {
            const int lr = wm * 32 + mt * 16 + r;
            a[mt].q[0] = *reinterpret_cast<const uint4*>(&As[buf][lr * 32 + kh]);
            a[mt].q[1] = *reinterpret_cast<const uint4*>(&As[buf][lr * 32 + 16 + kh]);
        }
        Frag bq[4];
        #pragma unroll
        for (int nt = 0; nt < 4; ++nt) {
            const int lc = wn * 64 + nt * 16 + r;
            bq[nt].q[0] = *reinterpret_cast<const uint4*>(&Bs[buf][lc * 32 + kh]);
            bq[nt].q[1] = *reinterpret_cast<const uint4*>(&Bs[buf][lc * 32 + 16 + kh]);
        }
        #pragma unroll
        for (int mt = 0; mt < 2; ++mt)
            #pragma unroll
            for (int nt = 0; nt < 4; ++nt)
                acc[mt][nt] = __builtin_amdgcn_wmma_f32_16x16x32_bf16(
                    false, a[mt].v, false, bq[nt].v, (short)0, acc[mt][nt],
                    false, false);
        __syncthreads();
    }

    // epilogue: C/D layout -> row = 8*(lane>=16)+i, col = lane&15
    const int rbase = (lane >> 4) * 8;
    const int cloc  = lane & 15;
    #pragma unroll
    for (int mt = 0; mt < 2; ++mt) {
        #pragma unroll
        for (int nt = 0; nt < 4; ++nt) {
            const int gc = col0 + wn * 64 + nt * 16 + cloc;
            const float bv = bias[gc];
            #pragma unroll
            for (int i = 0; i < 8; ++i) {
                const int gr = row0 + wm * 32 + mt * 16 + rbase + i;
                float v = acc[mt][nt][i] + bv;
                if (GELU) v = gelu_f(v);
                if (RES)  v += residual[(size_t)gr * N + gc];
                if (WF) outf[(size_t)gr * N + gc] = v;
                if (WB) outb[(size_t)gr * N + gc] = (bf16)v;
            }
        }
    }
}

// ---------------------------------------------------------------------------
// ox = sw[B,H,N,G] @ os[B,H,G,D]  (WMMA, K=64), written as bf16 [B,N,C]
// grid: (N/128, B*H); 8 waves each own 16 rows x 32 cols.
// ---------------------------------------------------------------------------
__global__ __launch_bounds__(256)
void ox_gemm_kernel(const bf16* __restrict__ sw, const bf16* __restrict__ osb,
                    bf16* __restrict__ oxb)
{
    __shared__ bf16 Bt[DD * GG];           // [d][g]
    const int tid  = threadIdx.x;
    const int lane = tid & 31;
    const int wave = tid >> 5;
    const int bh = blockIdx.y;
    const int b = bh >> 3, h = bh & 7;
    const size_t swbase = (size_t)bh * NN * GG;

    { // stage os transposed
        const int g  = tid >> 2;
        const int d0 = (tid & 3) * 8;
        #pragma unroll
        for (int j = 0; j < 8; ++j)
            Bt[(d0 + j) * GG + g] = osb[(size_t)bh * GG * DD + g * DD + d0 + j];
    }
    __syncthreads();

    const int r  = lane & 15;
    const int kh = (lane >> 4) * 8;
    const int tok0 = blockIdx.x * 128 + wave * 16;

    v8f acc[2];
    acc[0] = (v8f){0.f,0.f,0.f,0.f,0.f,0.f,0.f,0.f};
    acc[1] = acc[0];

    #pragma unroll
    for (int k0 = 0; k0 < GG; k0 += 32) {
        Frag a;
        const bf16* ap = sw + swbase + (size_t)(tok0 + r) * GG + k0;
        a.q[0] = *reinterpret_cast<const uint4*>(ap + kh);
        a.q[1] = *reinterpret_cast<const uint4*>(ap + 16 + kh);
        #pragma unroll
        for (int nt = 0; nt < 2; ++nt) {
            Frag bq;
            const int d  = nt * 16 + r;
            const int gb = k0 + (lane >> 4) * 8;
            bq.q[0] = *reinterpret_cast<const uint4*>(&Bt[d * GG + gb]);
            bq.q[1] = *reinterpret_cast<const uint4*>(&Bt[d * GG + gb + 16]);
            acc[nt] = __builtin_amdgcn_wmma_f32_16x16x32_bf16(
                false, a.v, false, bq.v, (short)0, acc[nt], false, false);
        }
    }
    const int rbase = (lane >> 4) * 8;
    #pragma unroll
    for (int nt = 0; nt < 2; ++nt) {
        const int d = nt * 16 + (lane & 15);
        #pragma unroll
        for (int i = 0; i < 8; ++i) {
            const int tok = tok0 + rbase + i;
            oxb[((size_t)b * NN + tok) * CC + h * DD + d] = (bf16)acc[nt][i];
        }
    }
}

// ---------------------------------------------------------------------------
// Slice logits + softmax: one wave per (b,n,h); lane owns g and g+32.
// ---------------------------------------------------------------------------
__global__ __launch_bounds__(256)
void slice_softmax_kernel(const float* __restrict__ xm, const float* __restrict__ wslice,
                          const float* __restrict__ bslice, const float* __restrict__ temp,
                          bf16* __restrict__ sw)
{
    const int lane = threadIdx.x & 31;
    const int wave = threadIdx.x >> 5;
    const int p  = blockIdx.x * 8 + wave;      // (b*N + n)*H + h
    const int h  = p & 7;
    const int bn = p >> 3;                      // b*N + n
    const int b  = bn >> 13;                    // N = 8192
    const int n  = bn & (NN - 1);

    const float* xp = xm + (size_t)bn * CC + h * DD;
    const int ga = lane, gb = lane + 32;
    float la = bslice[ga], lb = bslice[gb];
    #pragma unroll
    for (int d = 0; d < DD; ++d) {
        const float xv = xp[d];
        la += xv * wslice[d * GG + ga];
        lb += xv * wslice[d * GG + gb];
    }
    const float it = 1.0f / temp[h];
    la *= it; lb *= it;
    float m = fmaxf(la, lb);
    #pragma unroll
    for (int o = 16; o > 0; o >>= 1) m = fmaxf(m, __shfl_xor(m, o, 32));
    float ea = expf(la - m), eb = expf(lb - m);
    float s = ea + eb;
    #pragma unroll
    for (int o = 16; o > 0; o >>= 1) s += __shfl_xor(s, o, 32);
    const float inv = 1.0f / s;
    bf16* op = sw + ((size_t)(b * HH + h) * NN + n) * GG;
    op[ga] = (bf16)(ea * inv);
    op[gb] = (bf16)(eb * inv);
}

// ---------------------------------------------------------------------------
// Fused slice-token attention: stok/snorm reduction over N, qkv, 64x64
// attention, os.  One block per (b,h).
// ---------------------------------------------------------------------------
struct LoadBuf { bf16 swc[64 * 64]; float fxc[64 * 32]; };
union  SmemU   { LoadBuf ld; float scores[64 * 64]; };

__global__ __launch_bounds__(256)
void slice_attn_kernel(const bf16* __restrict__ sw, const float* __restrict__ fxm,
                       const float* __restrict__ wq, const float* __restrict__ wk,
                       const float* __restrict__ wv, bf16* __restrict__ osb)
{
    __shared__ SmemU u;
    __shared__ float stok_s[GG * DD];
    __shared__ float snorm_s[GG];
    __shared__ float q_s[GG * DD], k_s[GG * DD], v_s[GG * DD];

    const int tid = threadIdx.x;
    const int bh  = blockIdx.x;
    const int b = bh >> 3, h = bh & 7;
    const int g   = tid >> 2;        // 0..63
    const int seg = tid & 3;         // 0..3
    const int d0  = seg * 8;

    float acc[8] = {0.f,0.f,0.f,0.f,0.f,0.f,0.f,0.f};
    float sn = 0.f;
    const size_t swbase = (size_t)bh * NN * GG;

    for (int c = 0; c < NN / 64; ++c) {
        const int n0 = c * 64;
        {   // sw chunk [64 tok][64 g]
            const int tok = tid >> 2;
            const uint4* gp = reinterpret_cast<const uint4*>(
                sw + swbase + (size_t)(n0 + tok) * GG + seg * 16);
            uint4* sp = reinterpret_cast<uint4*>(&u.ld.swc[tok * 64 + seg * 16]);
            sp[0] = gp[0]; sp[1] = gp[1];
        }
        {   // fxm chunk [64 tok][32 d]
            const int tok = tid >> 2;
            const float4* gp = reinterpret_cast<const float4*>(
                fxm + ((size_t)b * NN + n0 + tok) * CC + h * DD + seg * 8);
            float4* sp = reinterpret_cast<float4*>(&u.ld.fxc[tok * 32 + seg * 8]);
            sp[0] = gp[0]; sp[1] = gp[1];
        }
        __syncthreads();
        for (int t = 0; t < 64; ++t) {
            const float s = (float)u.ld.swc[t * 64 + g];
            if (seg == 0) sn += s;
            #pragma unroll
            for (int j = 0; j < 8; ++j) acc[j] += s * u.ld.fxc[t * 32 + d0 + j];
        }
        __syncthreads();
    }
    if (seg == 0) snorm_s[g] = sn;
    __syncthreads();
    const float inv = 1.0f / (snorm_s[g] + 1e-5f);
    #pragma unroll
    for (int j = 0; j < 8; ++j) stok_s[g * DD + d0 + j] = acc[j] * inv;
    __syncthreads();

    // q,k,v projections (D x D each)
    #pragma unroll
    for (int j = 0; j < 8; ++j) {
        const int d = d0 + j;
        float qv = 0.f, kv = 0.f, vv = 0.f;
        for (int e = 0; e < DD; ++e) {
            const float sv = stok_s[g * DD + e];
            qv += sv * wq[e * DD + d];
            kv += sv * wk[e * DD + d];
            vv += sv * wv[e * DD + d];
        }
        q_s[g * DD + d] = qv; k_s[g * DD + d] = kv; v_s[g * DD + d] = vv;
    }
    __syncthreads();

    const float scale = 0.17677669529663687f;   // D^-0.5
    for (int j = 0; j < 16; ++j) {
        const int kk = seg * 16 + j;
        float sc = 0.f;
        for (int d = 0; d < DD; ++d) sc += q_s[g * DD + d] * k_s[kk * DD + d];
        u.scores[g * 64 + kk] = sc * scale;
    }
    __syncthreads();
    if (tid < 64) {
        float m = -1e30f;
        for (int k2 = 0; k2 < 64; ++k2) m = fmaxf(m, u.scores[tid * 64 + k2]);
        float s2 = 0.f;
        for (int k2 = 0; k2 < 64; ++k2) {
            float e = expf(u.scores[tid * 64 + k2] - m);
            u.scores[tid * 64 + k2] = e; s2 += e;
        }
        const float iv = 1.0f / s2;
        for (int k2 = 0; k2 < 64; ++k2) u.scores[tid * 64 + k2] *= iv;
    }
    __syncthreads();
    #pragma unroll
    for (int j = 0; j < 8; ++j) {
        const int d = d0 + j;
        float o = 0.f;
        for (int k2 = 0; k2 < 64; ++k2) o += u.scores[g * 64 + k2] * v_s[k2 * DD + d];
        osb[(size_t)bh * GG * DD + g * DD + d] = (bf16)o;
    }
}

// ---------------------------------------------------------------------------
// LayerNorm (C=256) -> bf16 output; one block per row.
// ---------------------------------------------------------------------------
__global__ __launch_bounds__(256)
void ln_bf16_kernel(const float* __restrict__ in, const float* __restrict__ gw,
                    const float* __restrict__ bw, bf16* __restrict__ outb)
{
    __shared__ float red[256];
    const int row = blockIdx.x, c = threadIdx.x;
    const float v = in[(size_t)row * CC + c];
    red[c] = v; __syncthreads();
    for (int o = 128; o > 0; o >>= 1) { if (c < o) red[c] += red[c + o]; __syncthreads(); }
    const float mean = red[0] / (float)CC; __syncthreads();
    const float dv = v - mean;
    red[c] = dv * dv; __syncthreads();
    for (int o = 128; o > 0; o >>= 1) { if (c < o) red[c] += red[c + o]; __syncthreads(); }
    const float rs = rsqrtf(red[0] / (float)CC + 1e-5f);
    outb[(size_t)row * CC + c] = (bf16)(dv * rs * gw[c] + bw[c]);
}

// ---------------------------------------------------------------------------
// Final: LN3 + projection to 1 output per token.
// ---------------------------------------------------------------------------
__global__ __launch_bounds__(256)
void final_kernel(const float* __restrict__ fx, const float* __restrict__ g3,
                  const float* __restrict__ b3, const float* __restrict__ ow,
                  const float* __restrict__ ob, float* __restrict__ out)
{
    __shared__ float red[256];
    const int row = blockIdx.x, c = threadIdx.x;
    const float v = fx[(size_t)row * CC + c];
    red[c] = v; __syncthreads();
    for (int o = 128; o > 0; o >>= 1) { if (c < o) red[c] += red[c + o]; __syncthreads(); }
    const float mean = red[0] / (float)CC; __syncthreads();
    const float dv = v - mean;
    red[c] = dv * dv; __syncthreads();
    for (int o = 128; o > 0; o >>= 1) { if (c < o) red[c] += red[c + o]; __syncthreads(); }
    const float rs = rsqrtf(red[0] / (float)CC + 1e-5f); __syncthreads();
    red[c] = (dv * rs * g3[c] + b3[c]) * ow[c]; __syncthreads();
    for (int o = 128; o > 0; o >>= 1) { if (c < o) red[c] += red[c + o]; __syncthreads(); }
    if (c == 0) out[row] = red[0] + ob[0];
}

// ---------------------------------------------------------------------------
// Preprocess MLP: fx = gelu([x,fx_in] @ w1 + b1) @ w2 + b2 + placeholder
// ---------------------------------------------------------------------------
__global__ __launch_bounds__(256)
void pre_kernel(const float* __restrict__ x, const float* __restrict__ fxin,
                const float* __restrict__ w1, const float* __restrict__ b1,
                const float* __restrict__ w2, const float* __restrict__ b2,
                const float* __restrict__ ph, float* __restrict__ fx)
{
    __shared__ float h1[512];
    __shared__ float in6[6];
    const int tok = blockIdx.x, tid = threadIdx.x;
    if (tid < 3) { in6[tid] = x[tok * 3 + tid]; in6[tid + 3] = fxin[tok * 3 + tid]; }
    __syncthreads();
    #pragma unroll
    for (int rep = 0; rep < 2; ++rep) {
        const int j = tid + rep * 256;
        float s = b1[j];
        #pragma unroll
        for (int i = 0; i < 6; ++i) s += in6[i] * w1[i * 512 + j];
        h1[j] = gelu_f(s);
    }
    __syncthreads();
    float s = b2[tid];
    for (int j = 0; j < 512; ++j) s += h1[j] * w2[j * 256 + tid];
    fx[(size_t)tok * CC + tid] = s + ph[tid];
}

// f32 [L,K,N] -> bf16 transposed [L,N,K]
__global__ void convT_bf16_kernel(const float* __restrict__ in, bf16* __restrict__ out,
                                  int K, int N)
{
    const int l = blockIdx.y;
    const int i = blockIdx.x * 256 + threadIdx.x;       // over N*K
    if (i < N * K) {
        const int n = i / K, k = i - n * K;
        out[(size_t)l * N * K + i] = (bf16)in[(size_t)l * K * N + (size_t)k * N + n];
    }
}

// ---------------------------------------------------------------------------
extern "C" void kernel_launch(void* const* d_in, const int* in_sizes, int n_in,
                              void* d_out, int out_size, void* d_ws, size_t ws_size,
                              hipStream_t stream)
{
    (void)in_sizes; (void)n_in; (void)out_size; (void)ws_size;
    const float* x      = (const float*)d_in[0];
    const float* fx_in  = (const float*)d_in[1];
    const float* pre_w1 = (const float*)d_in[2];
    const float* pre_b1 = (const float*)d_in[3];
    const float* pre_w2 = (const float*)d_in[4];
    const float* pre_b2 = (const float*)d_in[5];
    const float* ph     = (const float*)d_in[6];
    const float* ln1_g  = (const float*)d_in[7];
    const float* ln1_b  = (const float*)d_in[8];
    const float* wfx    = (const float*)d_in[9];
    const float* bfx    = (const float*)d_in[10];
    const float* wx     = (const float*)d_in[11];
    const float* bx     = (const float*)d_in[12];
    const float* wslice = (const float*)d_in[13];
    const float* bslice = (const float*)d_in[14];
    const float* temp   = (const float*)d_in[15];
    const float* wq     = (const float*)d_in[16];
    const float* wk     = (const float*)d_in[17];
    const float* wv     = (const float*)d_in[18];
    const float* wo     = (const float*)d_in[19];
    const float* bo     = (const float*)d_in[20];
    const float* ln2_g  = (const float*)d_in[21];
    const float* ln2_b  = (const float*)d_in[22];
    const float* mw1    = (const float*)d_in[23];
    const float* mb1    = (const float*)d_in[24];
    const float* mw2    = (const float*)d_in[25];
    const float* mb2    = (const float*)d_in[26];
    const float* ln3_g  = (const float*)d_in[27];
    const float* ln3_b  = (const float*)d_in[28];
    const float* out_w  = (const float*)d_in[29];
    const float* out_b  = (const float*)d_in[30];
    float* out = (float*)d_out;

    char* ws = (char*)d_ws;
    size_t off = 0;
    auto alloc = [&](size_t bytes) -> char* {
        char* p = ws + off; off += (bytes + 255) & ~(size_t)255; return p;
    };
    float* fx   = (float*)alloc((size_t)MT * CC * 4);
    bf16*  hbf  = (bf16*) alloc((size_t)MT * CC * 2);
    float* fxm  = (float*)alloc((size_t)MT * CC * 4);
    float* xm   = (float*)alloc((size_t)MT * CC * 4);
    bf16*  swb  = (bf16*) alloc((size_t)BB * HH * NN * GG * 2);
    bf16*  osb  = (bf16*) alloc((size_t)BB * HH * GG * DD * 2);
    bf16*  oxb  = (bf16*) alloc((size_t)MT * CC * 2);
    bf16*  g1b  = (bf16*) alloc((size_t)MT * FF * 2);
    bf16*  wfxt = (bf16*) alloc((size_t)LL * CC * CC * 2);   // [l][N=256][K=256]
    bf16*  wxt  = (bf16*) alloc((size_t)LL * CC * CC * 2);
    bf16*  wot  = (bf16*) alloc((size_t)LL * CC * CC * 2);
    bf16*  mw1t = (bf16*) alloc((size_t)LL * CC * FF * 2);   // [l][N=1024][K=256]
    bf16*  mw2t = (bf16*) alloc((size_t)LL * FF * CC * 2);   // [l][N=256][K=1024]

    auto convT = [&](const float* s, bf16* d, int K, int N) {
        convT_bf16_kernel<<<dim3((N * K + 255) / 256, LL), 256, 0, stream>>>(s, d, K, N);
    };
    convT(wfx, wfxt, CC, CC);
    convT(wx,  wxt,  CC, CC);
    convT(wo,  wot,  CC, CC);
    convT(mw1, mw1t, CC, FF);
    convT(mw2, mw2t, FF, CC);

    pre_kernel<<<MT, 256, 0, stream>>>(x, fx_in, pre_w1, pre_b1, pre_w2, pre_b2, ph, fx);

    const dim3 gC(CC / 128, MT / 128);   // 2 x 256
    const dim3 gF(FF / 128, MT / 128);   // 8 x 256

    for (int l = 0; l < LL; ++l) {
        ln_bf16_kernel<<<MT, 256, 0, stream>>>(fx, ln1_g + l * CC, ln1_b + l * CC, hbf);
        // fxm = h @ wfx + bfx
        gemm_bf16_kernel<false,false,true,false><<<gC, 256, 0, stream>>>(
            hbf, wfxt + (size_t)l * CC * CC, bfx + l * CC, nullptr, fxm, nullptr, CC, CC);
        // xm = h @ wx + bx
        gemm_bf16_kernel<false,false,true,false><<<gC, 256, 0, stream>>>(
            hbf, wxt + (size_t)l * CC * CC, bx + l * CC, nullptr, xm, nullptr, CC, CC);
        slice_softmax_kernel<<<MT * HH / 8, 256, 0, stream>>>(
            xm, wslice + l * DD * GG, bslice + l * GG, temp + l * HH, swb);
        slice_attn_kernel<<<BB * HH, 256, 0, stream>>>(
            swb, fxm, wq + l * DD * DD, wk + l * DD * DD, wv + l * DD * DD, osb);
        ox_gemm_kernel<<<dim3(NN / 128, BB * HH), 256, 0, stream>>>(swb, osb, oxb);
        // fx = ox @ wo + bo + fx
        gemm_bf16_kernel<false,true,true,false><<<gC, 256, 0, stream>>>(
            oxb, wot + (size_t)l * CC * CC, bo + l * CC, fx, fx, nullptr, CC, CC);
        ln_bf16_kernel<<<MT, 256, 0, stream>>>(fx, ln2_g + l * CC, ln2_b + l * CC, hbf);
        // g1 = gelu(h2 @ w1 + b1)   (bf16 out)
        gemm_bf16_kernel<true,false,false,true><<<gF, 256, 0, stream>>>(
            hbf, mw1t + (size_t)l * CC * FF, mb1 + l * FF, nullptr, nullptr, g1b, FF, CC);
        // fx = g1 @ w2 + b2 + fx
        gemm_bf16_kernel<false,true,true,false><<<gC, 256, 0, stream>>>(
            g1b, mw2t + (size_t)l * FF * CC, mb2 + l * CC, fx, fx, nullptr, CC, FF);
    }
    final_kernel<<<MT, 256, 0, stream>>>(fx, ln3_g, ln3_b, out_w, out_b, out);
}
